// _NonLocalBlockND_81758997447128
// MI455X (gfx1250) — matrix operand order repaired and checked by reading, not compiled, not run on hardware
//
#include <hip/hip_runtime.h>

// ---------------------------------------------------------------------------
// NonLocal block for MI455X (gfx1250, wave32, WMMA bf16 16x16x32, f32 accum)
// Sizes hardcoded from reference: B=8, C1=C2=256, Ci=128, H=W=64
// ---------------------------------------------------------------------------

#define B_   8
#define C1_  256
#define CI_  128
#define H_   64
#define W_   64
#define N1_  4096   // H*W
#define N2_  1024   // (H/2)*(W/2)

typedef __attribute__((ext_vector_type(16))) __bf16 v16bf;
typedef __attribute__((ext_vector_type(8)))  float  v8f;

struct alignas(16) U4 { unsigned int a, b, c, d; };

union Frag {
    v16bf  v;
    __bf16 e[16];
    U4     q[2];
};

__device__ __forceinline__ v8f wmma_bf16(v16bf a, v16bf b, v8f c) {
    // D(16x16 f32) = A(16x32 bf16) * B(32x16 bf16) + C
    return __builtin_amdgcn_wmma_f32_16x16x32_bf16(false, a, false, b,
                                                   (short)0, c, false, false);
}

// ---------------------------------------------------------------------------
// K0: elementwise f32 -> bf16 (used once for the four weight matrices)
// ---------------------------------------------------------------------------
__global__ void __launch_bounds__(256)
cvt_bf16_kernel(const float* __restrict__ in, __bf16* __restrict__ outp, int n)
{
    int i = blockIdx.x * 256 + threadIdx.x;
    if (i < n) outp[i] = (__bf16)in[i];
}

// ---------------------------------------------------------------------------
// K1: theta projection.  theta[b,n,c] = sum_k x1[b,k,n]*theta_w[c,k] + b[c]
// Output bf16, layout (B, N1, Ci).  Block = 8 waves, each wave one c-tile.
// ---------------------------------------------------------------------------
__global__ void __launch_bounds__(256)
proj_theta_kernel(const float* __restrict__ x1, const __bf16* __restrict__ w,
                  const float* __restrict__ bias, __bf16* __restrict__ outp)
{
    __shared__ __attribute__((aligned(16))) __bf16 a_s[16][32];  // [m][k], bf16

    const int tid  = threadIdx.x;
    const int wid  = tid >> 5;
    const int lane = tid & 31;
    const int h    = lane >> 4;
    const int lm   = lane & 15;

    const int bi = blockIdx.x;              // 2048 = B * N1/16
    const int b  = bi >> 8;
    const int n0 = (bi & 255) << 4;
    const int c0 = wid << 4;                // 8 waves -> Ci=128

    v8f acc = (v8f)0.0f;

    for (int kc = 0; kc < C1_ / 32; ++kc) {
        const int k0 = kc * 32;
        __syncthreads();
        // cooperative coalesced load + one conversion per element (shared by 8 waves)
        #pragma unroll
        for (int it = 0; it < 2; ++it) {
            int idx = tid + it * 256;       // < 512
            int m = idx & 15, k = idx >> 4;
            a_s[m][k] = (__bf16)x1[((size_t)(b * C1_ + k0 + k)) * N1_ + n0 + m];
        }
        __syncthreads();

        Frag A;
        const U4* ap = (const U4*)&a_s[lm][0];   // row = 64B, 16B aligned chunks
        A.q[0] = ap[2 * h];                      // k = h*16 .. h*16+7
        A.q[1] = ap[2 * h + 1];                  // k = h*16+8 ..

        Frag Bv;
        const U4* wp = (const U4*)(w + (size_t)(c0 + lm) * C1_ + k0 + h * 16);
        Bv.q[0] = wp[0];
        Bv.q[1] = wp[1];

        acc = wmma_bf16(A.v, Bv.v, acc);
    }

    const float bv = bias[c0 + lm];
    #pragma unroll
    for (int i = 0; i < 8; ++i) {
        int n = n0 + i + 8 * h;
        outp[((size_t)(b * N1_ + n)) * CI_ + c0 + lm] = (__bf16)(acc[i] + bv);
    }
}

// ---------------------------------------------------------------------------
// K2: pooled projection (phi / g).  conv1x1 at the 4 pixels of each 2x2 pool
// window, then max.  transposed=1 -> layout (B,Ci,N2) (for g),
// transposed=0 -> layout (B,N2,Ci) (for phi).
// ---------------------------------------------------------------------------
__global__ void __launch_bounds__(256)
proj_pool_kernel(const float* __restrict__ x, const __bf16* __restrict__ w,
                 const float* __restrict__ bias, __bf16* __restrict__ outp,
                 int transposed)
{
    __shared__ __attribute__((aligned(16))) __bf16 xs[2][32][32]; // [dr][c2][k]

    const int tid  = threadIdx.x;
    const int wid  = tid >> 5;
    const int lane = tid & 31;
    const int h    = lane >> 4;
    const int lm   = lane & 15;

    const int bi   = blockIdx.x;            // 512 = B * N2/16
    const int b    = bi >> 6;
    const int m0   = (bi & 63) << 4;
    const int r    = m0 >> 5;               // pooled row
    const int q0   = m0 & 31;               // pooled col base (0 or 16)
    const int c0   = wid << 4;

    v8f acc4[4];
    #pragma unroll
    for (int p = 0; p < 4; ++p) acc4[p] = (v8f)0.0f;

    for (int kc = 0; kc < C1_ / 32; ++kc) {
        const int k0 = kc * 32;
        __syncthreads();
        // cooperative load: 32 k x (2 rows x 32 cols), coalesced along cols
        #pragma unroll
        for (int it = 0; it < 8; ++it) {
            int idx = tid + it * 256;       // < 2048
            int k = idx >> 6, rc = idx & 63, dr = rc >> 5, c2 = rc & 31;
            xs[dr][c2][k] = (__bf16)
                x[((size_t)(b * C1_ + k0 + k)) * N1_ + (2 * r + dr) * W_ + 2 * q0 + c2];
        }
        __syncthreads();

        Frag Bv;
        const U4* wp = (const U4*)(w + (size_t)(c0 + lm) * C1_ + k0 + h * 16);
        Bv.q[0] = wp[0];
        Bv.q[1] = wp[1];

        #pragma unroll
        for (int dr = 0; dr < 2; ++dr)
        #pragma unroll
        for (int dc = 0; dc < 2; ++dc) {
            Frag A;
            const U4* ap = (const U4*)&xs[dr][2 * lm + dc][0];
            A.q[0] = ap[2 * h];
            A.q[1] = ap[2 * h + 1];
            acc4[dr * 2 + dc] = wmma_bf16(A.v, Bv.v, acc4[dr * 2 + dc]);
        }
    }

    const float bv = bias[c0 + lm];
    #pragma unroll
    for (int i = 0; i < 8; ++i) {
        float v = fmaxf(fmaxf(acc4[0][i], acc4[1][i]),
                        fmaxf(acc4[2][i], acc4[3][i])) + bv;
        int m = m0 + i + 8 * h;
        int c = c0 + lm;
        size_t off = transposed ? ((size_t)(b * CI_ + c)) * N2_ + m
                                : ((size_t)(b * N2_ + m)) * CI_ + c;
        outp[off] = (__bf16)v;
    }
}

// ---------------------------------------------------------------------------
// K3: flash-attention core.  Each wave owns 16 query rows (full Ci=128 kept in
// registers) and streams phi/g in 32-key chunks with online softmax.
// ---------------------------------------------------------------------------
__global__ void __launch_bounds__(128)
attn_kernel(const __bf16* __restrict__ theta,   // (B,N1,Ci)
            const __bf16* __restrict__ phi,     // (B,N2,Ci)
            const __bf16* __restrict__ g,       // (B,Ci,N2)
            __bf16* __restrict__ y)             // (B,N1,Ci)
{
    __shared__ __attribute__((aligned(16))) __bf16 pbuf[4][16 * 32];

    const int wid  = threadIdx.x >> 5;
    const int lane = threadIdx.x & 31;
    const int h    = lane >> 4;
    const int lm   = lane & 15;

    const int gid = blockIdx.x * 4 + wid;   // < 2048 = B * N1/16
    const int b   = gid >> 8;
    const int n0  = (gid & 255) << 4;

    // theta A-fragments for all of Ci=128 (4 k-chunks of 32)
    Frag tA[4];
    {
        const U4* tp = (const U4*)(theta + ((size_t)b * N1_ + n0 + lm) * CI_);
        #pragma unroll
        for (int kc = 0; kc < 4; ++kc) {
            tA[kc].q[0] = tp[kc * 4 + 2 * h];
            tA[kc].q[1] = tp[kc * 4 + 2 * h + 1];
        }
    }

    v8f acc[8];
    #pragma unroll
    for (int ct = 0; ct < 8; ++ct) acc[ct] = (v8f)0.0f;
    float mrun[8], lrun[8];
    #pragma unroll
    for (int i = 0; i < 8; ++i) { mrun[i] = -__builtin_inff(); lrun[i] = 0.0f; }

    const U4* pp = (const U4*)(phi + (size_t)b * N2_ * CI_);
    const U4* gp = (const U4*)(g   + (size_t)b * CI_ * N2_);
    __bf16* pb = pbuf[wid];

    for (int m0 = 0; m0 < N2_; m0 += 32) {
        __builtin_prefetch(phi + ((size_t)b * N2_ + ((m0 + 32) & (N2_ - 1)) + lm) * CI_, 0, 1);

        // S = theta * phi^T : two 16x16 tiles over the 32 keys
        v8f s0 = (v8f)0.0f, s1 = (v8f)0.0f;
        #pragma unroll
        for (int kc = 0; kc < 4; ++kc) {
            Frag pB0, pB1;
            size_t r0 = ((size_t)(m0 + lm)) * CI_ + kc * 32 + h * 16;       // bf16 units
            size_t r1 = ((size_t)(m0 + 16 + lm)) * CI_ + kc * 32 + h * 16;
            pB0.q[0] = pp[r0 / 8]; pB0.q[1] = pp[r0 / 8 + 1];
            pB1.q[0] = pp[r1 / 8]; pB1.q[1] = pp[r1 / 8 + 1];
            s0 = wmma_bf16(tA[kc].v, pB0.v, s0);
            s1 = wmma_bf16(tA[kc].v, pB1.v, s1);
        }

        // online softmax (row = M dim = i + 8h; reduce across the 16-lane half)
        #pragma unroll
        for (int i = 0; i < 8; ++i) {
            float t = fmaxf(s0[i], s1[i]);
            t = fmaxf(t, __shfl_xor(t, 1, 16));
            t = fmaxf(t, __shfl_xor(t, 2, 16));
            t = fmaxf(t, __shfl_xor(t, 4, 16));
            t = fmaxf(t, __shfl_xor(t, 8, 16));
            float mnew = fmaxf(mrun[i], t);
            float corr = __expf(mrun[i] - mnew);
            mrun[i] = mnew;
            float p0 = __expf(s0[i] - mnew);
            float p1 = __expf(s1[i] - mnew);
            float rs = p0 + p1;
            rs += __shfl_xor(rs, 1, 16);
            rs += __shfl_xor(rs, 2, 16);
            rs += __shfl_xor(rs, 4, 16);
            rs += __shfl_xor(rs, 8, 16);
            lrun[i] = lrun[i] * corr + rs;
            #pragma unroll
            for (int ct = 0; ct < 8; ++ct) acc[ct][i] *= corr;
            int row = i + 8 * h;
            pb[row * 32 + lm]      = (__bf16)p0;  // D-layout -> LDS (row n, col m)
            pb[row * 32 + 16 + lm] = (__bf16)p1;
        }

        // re-read P from LDS in A-matrix layout (same-wave LDS is in-order)
        asm volatile("s_wait_dscnt 0" ::: "memory");
        Frag pA;
        const U4* pbq = (const U4*)pb;
        pA.q[0] = pbq[lm * 4 + 2 * h];
        pA.q[1] = pbq[lm * 4 + 2 * h + 1];
        asm volatile("" ::: "memory");

        // y += P(16x32) * G(32x128)
        #pragma unroll
        for (int ct = 0; ct < 8; ++ct) {
            Frag gB;
            size_t off = ((size_t)(ct * 16 + lm)) * N2_ + m0 + h * 16;      // bf16 units
            gB.q[0] = gp[off / 8]; gB.q[1] = gp[off / 8 + 1];
            acc[ct] = wmma_bf16(pA.v, gB.v, acc[ct]);
        }
    }

    #pragma unroll
    for (int i = 0; i < 8; ++i) {
        float inv = 1.0f / lrun[i];
        int n = n0 + i + 8 * h;
        #pragma unroll
        for (int ct = 0; ct < 8; ++ct)
            y[((size_t)(b * N1_ + n)) * CI_ + ct * 16 + lm] = (__bf16)(acc[ct][i] * inv);
    }
}

// ---------------------------------------------------------------------------
// K4: out = BN(conv1x1(y, W_w) + W_b) + x1
// ---------------------------------------------------------------------------
__global__ void __launch_bounds__(256)
final_kernel(const __bf16* __restrict__ y, const __bf16* __restrict__ Ww,
             const float* __restrict__ Wb, const float* __restrict__ gamma,
             const float* __restrict__ beta, const float* __restrict__ mean,
             const float* __restrict__ var, const float* __restrict__ x1,
             float* __restrict__ out)
{
    const int wid  = threadIdx.x >> 5;
    const int lane = threadIdx.x & 31;
    const int h    = lane >> 4;
    const int lm   = lane & 15;

    const int gid = blockIdx.x * 8 + wid;   // < 32768 = B * 16 * 256
    const int b   = gid >> 12;
    const int rem = gid & 4095;
    const int c0  = (rem >> 8) << 4;
    const int n0  = (rem & 255) << 4;

    v8f acc = (v8f)0.0f;
    const U4* yp = (const U4*)(y + ((size_t)b * N1_ + n0 + lm) * CI_);
    const U4* wp = (const U4*)(Ww + (size_t)(c0 + lm) * CI_ + h * 16);
    #pragma unroll
    for (int kc = 0; kc < 4; ++kc) {
        Frag A;
        A.q[0] = yp[kc * 4 + 2 * h];
        A.q[1] = yp[kc * 4 + 2 * h + 1];
        Frag Bv;
        Bv.q[0] = wp[kc * 4];       // kc*32 bf16 = kc*4 U4
        Bv.q[1] = wp[kc * 4 + 1];
        acc = wmma_bf16(A.v, Bv.v, acc);
    }

    const int co = c0 + lm;
    const float inv = gamma[co] * __frsqrt_rn(var[co] + 1e-5f);
    const float bb = beta[co], mu = mean[co], wb = Wb[co];

    const size_t base = ((size_t)(b * C1_ + co)) * N1_ + n0 + 8 * h;
    float4 r0 = ((const float4*)(x1 + base))[0];
    float4 r1 = ((const float4*)(x1 + base))[1];

    float o[8];
    #pragma unroll
    for (int i = 0; i < 8; ++i) o[i] = (acc[i] + wb - mu) * inv + bb;

    float4 w0 = make_float4(o[0] + r0.x, o[1] + r0.y, o[2] + r0.z, o[3] + r0.w);
    float4 w1 = make_float4(o[4] + r1.x, o[5] + r1.y, o[6] + r1.z, o[7] + r1.w);
    ((float4*)(out + base))[0] = w0;
    ((float4*)(out + base))[1] = w1;
}

// ---------------------------------------------------------------------------
extern "C" void kernel_launch(void* const* d_in, const int* in_sizes, int n_in,
                              void* d_out, int out_size, void* d_ws, size_t ws_size,
                              hipStream_t stream) {
    (void)in_sizes; (void)n_in; (void)out_size; (void)ws_size;
    const float* x1   = (const float*)d_in[0];
    const float* x2   = (const float*)d_in[1];
    const float* g_w  = (const float*)d_in[2];
    const float* g_b  = (const float*)d_in[3];
    const float* th_w = (const float*)d_in[4];
    const float* th_b = (const float*)d_in[5];
    const float* ph_w = (const float*)d_in[6];
    const float* ph_b = (const float*)d_in[7];
    const float* W_w  = (const float*)d_in[8];
    const float* W_b  = (const float*)d_in[9];
    const float* bng  = (const float*)d_in[10];
    const float* bnb  = (const float*)d_in[11];
    const float* bnm  = (const float*)d_in[12];
    const float* bnv  = (const float*)d_in[13];
    float* out = (float*)d_out;

    __bf16* theta_ws = (__bf16*)d_ws;                                 // (B,N1,Ci)
    __bf16* phi_ws   = theta_ws + (size_t)B_ * N1_ * CI_;             // (B,N2,Ci)
    __bf16* g_ws     = phi_ws   + (size_t)B_ * N2_ * CI_;             // (B,Ci,N2)
    __bf16* y_ws     = g_ws     + (size_t)B_ * N2_ * CI_;             // (B,N1,Ci)
    __bf16* thw_bf   = y_ws     + (size_t)B_ * N1_ * CI_;             // (Ci,C1)
    __bf16* gw_bf    = thw_bf   + (size_t)CI_ * C1_;
    __bf16* phw_bf   = gw_bf    + (size_t)CI_ * C1_;
    __bf16* Ww_bf    = phw_bf   + (size_t)CI_ * C1_;                  // (C1,Ci)

    const int NW = CI_ * C1_;   // 32768
    cvt_bf16_kernel<<<NW / 256, 256, 0, stream>>>(th_w, thw_bf, NW);
    cvt_bf16_kernel<<<NW / 256, 256, 0, stream>>>(g_w,  gw_bf,  NW);
    cvt_bf16_kernel<<<NW / 256, 256, 0, stream>>>(ph_w, phw_bf, NW);
    cvt_bf16_kernel<<<NW / 256, 256, 0, stream>>>(W_w,  Ww_bf,  NW);

    proj_theta_kernel<<<B_ * (N1_ / 16), 256, 0, stream>>>(x1, thw_bf, th_b, theta_ws);
    proj_pool_kernel <<<B_ * (N2_ / 16), 256, 0, stream>>>(x1, gw_bf, g_b, g_ws, 1);
    proj_pool_kernel <<<B_ * (N2_ / 16), 256, 0, stream>>>(x2, phw_bf, ph_b, phi_ws, 0);
    attn_kernel      <<<B_ * (N1_ / 16) / 4, 128, 0, stream>>>(theta_ws, phi_ws, g_ws, y_ws);
    final_kernel     <<<B_ * 16 * 256 / 8, 256, 0, stream>>>(y_ws, Ww_bf, W_b, bng, bnb,
                                                             bnm, bnv, x1, out);
}
